// MRDecoderS_84043920048420
// MI455X (gfx1250) — matrix operand order repaired and checked by reading, not compiled, not run on hardware
//
#include <hip/hip_runtime.h>
#include <hip/hip_bf16.h>
#include <math.h>

#define N_NODES 1024
#define T_LEN   16
#define N_EDGES 16384
#define HDIM    128
#define HDIM2   256
#define N_STEPS 15
#define KSPLIT  8
#define KCHUNK  (N_EDGES / KSPLIT)   // 2048

typedef __attribute__((ext_vector_type(16))) __bf16 v16bf;
typedef __attribute__((ext_vector_type(8)))  __bf16 v8bf;
typedef __attribute__((ext_vector_type(8)))  float  v8f;

__device__ __forceinline__ __bf16 f2bf(float f) {
  union { float f; unsigned u; } v; v.f = f;
  unsigned r = (v.u + 0x7FFFu + ((v.u >> 16) & 1u)) >> 16;
  unsigned short h = (unsigned short)r;
  __bf16 out; __builtin_memcpy(&out, &h, 2); return out;
}

__device__ __forceinline__ float sigm(float x) { return 1.0f / (1.0f + expf(-x)); }

// A fragment: 16x32 bf16, row-major source with leading dim ld.
__device__ __forceinline__ v16bf load_a_frag(const __bf16* A, int ld, int lane) {
  const int m = lane & 15, kh = (lane >> 4) & 1;
  const __bf16* p = A + (size_t)m * ld + kh * 8;
  union { v16bf v; v8bf h[2]; } u;
  u.h[0] = *(const v8bf*)(p);
  u.h[1] = *(const v8bf*)(p + 16);
  return u.v;
}
// B fragment from B-transposed (N x K row-major).
__device__ __forceinline__ v16bf load_bt_frag(const __bf16* BT, int ld, int lane) {
  const int n = lane & 15, kh = (lane >> 4) & 1;
  const __bf16* p = BT + (size_t)n * ld + kh * 8;
  union { v16bf v; v8bf h[2]; } u;
  u.h[0] = *(const v8bf*)(p);
  u.h[1] = *(const v8bf*)(p + 16);
  return u.v;
}
__device__ __forceinline__ v8f wmma_bf16(v16bf a, v16bf b, v8f c) {
  return __builtin_amdgcn_wmma_f32_16x16x32_bf16(false, a, false, b, (short)0, c, false, false);
}

// ---------------- prep kernels ----------------
__global__ void k_cvt(const float* __restrict__ src, __bf16* __restrict__ dst, int n) {
  int i = blockIdx.x * blockDim.x + threadIdx.x;
  if (i < n) dst[i] = f2bf(src[i]);
}

__global__ void k_transpose(const float* __restrict__ src, __bf16* __restrict__ dst,
                            int rows, int cols) {
  int i = blockIdx.x * blockDim.x + threadIdx.x;
  if (i < rows * cols) {
    int r = i / cols, c = i - r * cols;
    dst[(size_t)c * rows + r] = f2bf(src[i]);
  }
}

__global__ void k_init(float* __restrict__ hidden, __bf16* __restrict__ hidT,
                       float* __restrict__ prev_pred,
                       const float* __restrict__ cont,
                       const float* __restrict__ aW, const float* __restrict__ ab,
                       const float* __restrict__ gW, const float* __restrict__ gb,
                       float* __restrict__ scal) {
  int i = blockIdx.x * blockDim.x + threadIdx.x;
  if (i < N_NODES * HDIM) { hidden[i] = 0.0f; hidT[i] = f2bf(0.0f); }
  if (i < N_NODES) prev_pred[i] = 0.0f;
  if (i == 0) {
    float a = cont[0] * aW[0] + cont[1] * aW[1] + ab[0];
    float g = cont[0] * gW[0] + cont[1] * gW[1] + gb[0];
    scal[0] = fmaxf(a, 0.1f);
    scal[1] = fmaxf(g, 0.1f);
  }
}

// ---------------- per-step kernels ----------------
// msg_pre[:, 0:128] = sendR @ hidden ; msg_pre[:, 128:256] = recvR @ hidden
// one wave per 16x64 output tile; ping-pong fragment sets, unroll-by-2 in K.
__global__ __launch_bounds__(32) void k_msgpre(
    const __bf16* __restrict__ sendR, const __bf16* __restrict__ recvR,
    const __bf16* __restrict__ hidT, __bf16* __restrict__ msg_pre) {
  const int lane = threadIdx.x;
  const int bx = blockIdx.x;                 // 0..3: {sendR lo, sendR hi, recvR lo, recvR hi}
  const int e0 = blockIdx.y * 16;
  const __bf16* A = (bx < 2 ? sendR : recvR) + (size_t)e0 * N_NODES;
  const int hbase = (bx & 1) * 64;
  const int colbase = (bx < 2 ? 0 : 128) + hbase;
  const __bf16* BT0 = hidT + (size_t)(hbase + 0)  * N_NODES;
  const __bf16* BT1 = hidT + (size_t)(hbase + 16) * N_NODES;
  const __bf16* BT2 = hidT + (size_t)(hbase + 32) * N_NODES;
  const __bf16* BT3 = hidT + (size_t)(hbase + 48) * N_NODES;
  v8f c0 = {}, c1 = {}, c2 = {}, c3 = {};
  // p-set holds fragments for k0; q-set for k0+32
  v16bf ap  = load_a_frag(A, N_NODES, lane);
  v16bf bp0 = load_bt_frag(BT0, N_NODES, lane);
  v16bf bp1 = load_bt_frag(BT1, N_NODES, lane);
  v16bf bp2 = load_bt_frag(BT2, N_NODES, lane);
  v16bf bp3 = load_bt_frag(BT3, N_NODES, lane);
#pragma unroll 1
  for (int k0 = 0; k0 < N_NODES - 64; k0 += 64) {
    __builtin_prefetch((const char*)(A + (size_t)(lane & 15) * N_NODES + k0 + 256), 0, 3);
    v16bf aq  = load_a_frag(A + k0 + 32, N_NODES, lane);
    v16bf bq0 = load_bt_frag(BT0 + k0 + 32, N_NODES, lane);
    v16bf bq1 = load_bt_frag(BT1 + k0 + 32, N_NODES, lane);
    v16bf bq2 = load_bt_frag(BT2 + k0 + 32, N_NODES, lane);
    v16bf bq3 = load_bt_frag(BT3 + k0 + 32, N_NODES, lane);
    c0 = wmma_bf16(ap, bp0, c0);
    c1 = wmma_bf16(ap, bp1, c1);
    c2 = wmma_bf16(ap, bp2, c2);
    c3 = wmma_bf16(ap, bp3, c3);
    ap  = load_a_frag(A + k0 + 64, N_NODES, lane);
    bp0 = load_bt_frag(BT0 + k0 + 64, N_NODES, lane);
    bp1 = load_bt_frag(BT1 + k0 + 64, N_NODES, lane);
    bp2 = load_bt_frag(BT2 + k0 + 64, N_NODES, lane);
    bp3 = load_bt_frag(BT3 + k0 + 64, N_NODES, lane);
    c0 = wmma_bf16(aq, bq0, c0);
    c1 = wmma_bf16(aq, bq1, c1);
    c2 = wmma_bf16(aq, bq2, c2);
    c3 = wmma_bf16(aq, bq3, c3);
  }
  {   // tail: p-set @ N-64, q-set @ N-32
    v16bf aq  = load_a_frag(A + N_NODES - 32, N_NODES, lane);
    v16bf bq0 = load_bt_frag(BT0 + N_NODES - 32, N_NODES, lane);
    v16bf bq1 = load_bt_frag(BT1 + N_NODES - 32, N_NODES, lane);
    v16bf bq2 = load_bt_frag(BT2 + N_NODES - 32, N_NODES, lane);
    v16bf bq3 = load_bt_frag(BT3 + N_NODES - 32, N_NODES, lane);
    c0 = wmma_bf16(ap, bp0, c0);
    c1 = wmma_bf16(ap, bp1, c1);
    c2 = wmma_bf16(ap, bp2, c2);
    c3 = wmma_bf16(ap, bp3, c3);
    c0 = wmma_bf16(aq, bq0, c0);
    c1 = wmma_bf16(aq, bq1, c1);
    c2 = wmma_bf16(aq, bq2, c2);
    c3 = wmma_bf16(aq, bq3, c3);
  }
  const int n = lane & 15, kh = (lane >> 4) & 1;
#pragma unroll
  for (int v = 0; v < 8; ++v) {
    size_t row = (size_t)(e0 + v + 8 * kh) * HDIM2 + colbase + n;
    msg_pre[row + 0]  = f2bf(c0[v]);
    msg_pre[row + 16] = f2bf(c1[v]);
    msg_pre[row + 32] = f2bf(c2[v]);
    msg_pre[row + 48] = f2bf(c3[v]);
  }
}

// msgc^T = ( tanh( tanh(msg_pre @ W1^T + b1) @ W2^T + b2 ) * edges[:,1] )^T  (H x E)
__global__ __launch_bounds__(256) void k_mlp(
    const __bf16* __restrict__ msg_pre, const __bf16* __restrict__ W1,
    const __bf16* __restrict__ W2, const float* __restrict__ b1,
    const float* __restrict__ b2, const float* __restrict__ edges,
    __bf16* __restrict__ msgcT) {
  __shared__ __bf16 m1s[16 * HDIM];
  const int lane = threadIdx.x & 31;
  const int w = threadIdx.x >> 5;
  const int e0 = blockIdx.x * 16;
  const int h0 = w * 16;
  const int n = lane & 15, kh = (lane >> 4) & 1;
  {   // layer 1
    v8f c = {};
    const __bf16* A  = msg_pre + (size_t)e0 * HDIM2;
    const __bf16* BT = W1 + (size_t)h0 * HDIM2;
#pragma unroll
    for (int k0 = 0; k0 < HDIM2; k0 += 32)
      c = wmma_bf16(load_a_frag(A + k0, HDIM2, lane),
                    load_bt_frag(BT + k0, HDIM2, lane), c);
#pragma unroll
    for (int v = 0; v < 8; ++v)
      m1s[(v + 8 * kh) * HDIM + h0 + n] = f2bf(tanhf(c[v] + b1[h0 + n]));
  }
  __syncthreads();
  {   // layer 2 + edge gating, store transposed
    v8f c = {};
    const __bf16* BT = W2 + (size_t)h0 * HDIM;
#pragma unroll
    for (int k0 = 0; k0 < HDIM; k0 += 32)
      c = wmma_bf16(load_a_frag(&m1s[k0], HDIM, lane),
                    load_bt_frag(BT + k0, HDIM, lane), c);
#pragma unroll
    for (int v = 0; v < 8; ++v) {
      int e = e0 + v + 8 * kh;
      float ew = edges[(size_t)e * 2 + 1];
      msgcT[(size_t)(h0 + n) * N_EDGES + e] = f2bf(tanhf(c[v] + b2[h0 + n]) * ew);
    }
  }
}

// Split-K aggregation: partial[ks] = recvRT[:, ksrange] @ msgc[ksrange, :]
// one wave per 32x32 output tile per K chunk; ping-pong fragment sets.
__global__ __launch_bounds__(32) void k_agg(
    const __bf16* __restrict__ recvRT, const __bf16* __restrict__ msgcT,
    float* __restrict__ part) {
  const int lane = threadIdx.x;
  const int h0 = blockIdx.x * 32;
  const int n0 = blockIdx.y * 32;
  const int ks = blockIdx.z;
  const int kbeg = ks * KCHUNK, kend = kbeg + KCHUNK;
  const __bf16* A0 = recvRT + (size_t)n0 * N_EDGES;
  const __bf16* A1 = recvRT + (size_t)(n0 + 16) * N_EDGES;
  const __bf16* B0 = msgcT + (size_t)h0 * N_EDGES;
  const __bf16* B1 = msgcT + (size_t)(h0 + 16) * N_EDGES;
  v8f c00 = {}, c01 = {}, c10 = {}, c11 = {};
  v16bf a0p = load_a_frag(A0 + kbeg, N_EDGES, lane);
  v16bf a1p = load_a_frag(A1 + kbeg, N_EDGES, lane);
  v16bf b0p = load_bt_frag(B0 + kbeg, N_EDGES, lane);
  v16bf b1p = load_bt_frag(B1 + kbeg, N_EDGES, lane);
#pragma unroll 1
  for (int k0 = kbeg; k0 < kend - 64; k0 += 64) {
    __builtin_prefetch((const char*)(A0 + (size_t)(lane & 15) * N_EDGES + k0 + 512), 0, 3);
    v16bf a0q = load_a_frag(A0 + k0 + 32, N_EDGES, lane);
    v16bf a1q = load_a_frag(A1 + k0 + 32, N_EDGES, lane);
    v16bf b0q = load_bt_frag(B0 + k0 + 32, N_EDGES, lane);
    v16bf b1q = load_bt_frag(B1 + k0 + 32, N_EDGES, lane);
    c00 = wmma_bf16(a0p, b0p, c00);
    c01 = wmma_bf16(a0p, b1p, c01);
    c10 = wmma_bf16(a1p, b0p, c10);
    c11 = wmma_bf16(a1p, b1p, c11);
    a0p = load_a_frag(A0 + k0 + 64, N_EDGES, lane);
    a1p = load_a_frag(A1 + k0 + 64, N_EDGES, lane);
    b0p = load_bt_frag(B0 + k0 + 64, N_EDGES, lane);
    b1p = load_bt_frag(B1 + k0 + 64, N_EDGES, lane);
    c00 = wmma_bf16(a0q, b0q, c00);
    c01 = wmma_bf16(a0q, b1q, c01);
    c10 = wmma_bf16(a1q, b0q, c10);
    c11 = wmma_bf16(a1q, b1q, c11);
  }
  {   // tail: p-set @ kend-64, q-set @ kend-32
    v16bf a0q = load_a_frag(A0 + kend - 32, N_EDGES, lane);
    v16bf a1q = load_a_frag(A1 + kend - 32, N_EDGES, lane);
    v16bf b0q = load_bt_frag(B0 + kend - 32, N_EDGES, lane);
    v16bf b1q = load_bt_frag(B1 + kend - 32, N_EDGES, lane);
    c00 = wmma_bf16(a0p, b0p, c00);
    c01 = wmma_bf16(a0p, b1p, c01);
    c10 = wmma_bf16(a1p, b0p, c10);
    c11 = wmma_bf16(a1p, b1p, c11);
    c00 = wmma_bf16(a0q, b0q, c00);
    c01 = wmma_bf16(a0q, b1q, c01);
    c10 = wmma_bf16(a1q, b0q, c10);
    c11 = wmma_bf16(a1q, b1q, c11);
  }
  float* P = part + (size_t)ks * N_NODES * HDIM;
  const int n = lane & 15, kh = (lane >> 4) & 1;
#pragma unroll
  for (int v = 0; v < 8; ++v) {
    int r = v + 8 * kh;
    P[(size_t)(n0 + r)      * HDIM + h0 + n]      = c00[v];
    P[(size_t)(n0 + r)      * HDIM + h0 + 16 + n] = c01[v];
    P[(size_t)(n0 + 16 + r) * HDIM + h0 + n]      = c10[v];
    P[(size_t)(n0 + 16 + r) * HDIM + h0 + 16 + n] = c11[v];
  }
}

// Deterministic reduction of K-split partials, apply 1/N, emit bf16 msg_agg.
__global__ void k_agg_reduce(const float* __restrict__ part, __bf16* __restrict__ maggbf) {
  int i = blockIdx.x * blockDim.x + threadIdx.x;
  if (i < N_NODES * HDIM) {
    float s = 0.0f;
#pragma unroll
    for (int ks = 0; ks < KSPLIT; ++ks) s += part[(size_t)ks * N_NODES * HDIM + i];
    maggbf[i] = f2bf(s * (1.0f / (float)N_NODES));
  }
}

// GRU update + 3-layer output MLP + saturation on last node, per 16-node strip.
__global__ __launch_bounds__(256) void k_gru_out(
    const __bf16* __restrict__ maggbf,
    const __bf16* __restrict__ hidRbf, const __bf16* __restrict__ hidIbf,
    const __bf16* __restrict__ hidNbf,
    const float* __restrict__ inR, const float* __restrict__ inI,
    const float* __restrict__ inN,
    const __bf16* __restrict__ o1Wbf, const float* __restrict__ o1b,
    const __bf16* __restrict__ o2Wbf, const float* __restrict__ o2b,
    const float* __restrict__ o3W, const float* __restrict__ o3b,
    const float* __restrict__ data, const int* __restrict__ tml,
    float* __restrict__ hidden, __bf16* __restrict__ hidT,
    float* __restrict__ prev_pred, const float* __restrict__ scal,
    float* __restrict__ out, int t) {
  __shared__ __bf16 hs[16 * HDIM];
  __shared__ __bf16 p1s[16 * HDIM];
  __shared__ float  p2s[16 * HDIM];
  const int lane = threadIdx.x & 31;
  const int w = threadIdx.x >> 5;
  const int n0 = blockIdx.x * 16;
  const int h0 = w * 16;
  const int nl = lane & 15, kh = (lane >> 4) & 1;

  v8f cR = {}, cI = {}, cN = {};
  {
    const __bf16* A = maggbf + (size_t)n0 * HDIM;
#pragma unroll
    for (int k0 = 0; k0 < HDIM; k0 += 32) {
      v16bf a = load_a_frag(A + k0, HDIM, lane);
      cR = wmma_bf16(a, load_bt_frag(hidRbf + (size_t)h0 * HDIM + k0, HDIM, lane), cR);
      cI = wmma_bf16(a, load_bt_frag(hidIbf + (size_t)h0 * HDIM + k0, HDIM, lane), cI);
      cN = wmma_bf16(a, load_bt_frag(hidNbf + (size_t)h0 * HDIM + k0, HDIM, lane), cN);
    }
  }
  const int TML = *tml;
#pragma unroll
  for (int v = 0; v < 8; ++v) {
    int r = v + 8 * kh;
    int node = n0 + r;
    int h = h0 + nl;
    float x = (t <= TML) ? data[(size_t)node * T_LEN + t] : prev_pred[node];
    float rg = sigm(x * inR[h] + cR[v]);
    float ig = sigm(x * inI[h] + cI[v]);
    float nn = sigm(x * inN[h] + rg * cN[v]);
    float hold = hidden[(size_t)node * HDIM + h];
    float hnew = (1.0f - ig) * nn + ig * hold;
    hidden[(size_t)node * HDIM + h] = hnew;
    hidT[(size_t)h * N_NODES + node] = f2bf(hnew);
    hs[r * HDIM + h] = f2bf(hnew);
  }
  __syncthreads();
  {   // o1
    v8f c = {};
#pragma unroll
    for (int k0 = 0; k0 < HDIM; k0 += 32)
      c = wmma_bf16(load_a_frag(&hs[k0], HDIM, lane),
                    load_bt_frag(o1Wbf + (size_t)h0 * HDIM + k0, HDIM, lane), c);
#pragma unroll
    for (int v = 0; v < 8; ++v)
      p1s[(v + 8 * kh) * HDIM + h0 + nl] = f2bf(fmaxf(c[v] + o1b[h0 + nl], 0.0f));
  }
  __syncthreads();
  {   // o2
    v8f c = {};
#pragma unroll
    for (int k0 = 0; k0 < HDIM; k0 += 32)
      c = wmma_bf16(load_a_frag(&p1s[k0], HDIM, lane),
                    load_bt_frag(o2Wbf + (size_t)h0 * HDIM + k0, HDIM, lane), c);
#pragma unroll
    for (int v = 0; v < 8; ++v)
      p2s[(v + 8 * kh) * HDIM + h0 + nl] = fmaxf(c[v] + o2b[h0 + nl], 0.0f);
  }
  __syncthreads();
  if (threadIdx.x < 16) {
    int r = threadIdx.x;
    int node = n0 + r;
    float s = o3b[0];
    for (int k = 0; k < HDIM; ++k) s += p2s[r * HDIM + k] * o3W[k];
    if (node == N_NODES - 1) {
      float av = scal[0], gv = scal[1];
      float force = fmaxf(s, 0.1f);
      float fa = powf(force, av);
      s = fa / (fa + powf(gv, av)) * s;
    }
    prev_pred[node] = s;
    out[(size_t)node * N_STEPS + t] = s;
  }
}

// ---------------- host ----------------
extern "C" void kernel_launch(void* const* d_in, const int* in_sizes, int n_in,
                              void* d_out, int out_size, void* d_ws, size_t ws_size,
                              hipStream_t stream) {
  const float* data  = (const float*)d_in[0];
  const float* cont  = (const float*)d_in[1];
  const float* edges = (const float*)d_in[2];
  const float* recvR = (const float*)d_in[3];
  const float* sendR = (const float*)d_in[4];
  const int*   tml   = (const int*)d_in[6];
  const float* W1    = (const float*)d_in[7];
  const float* b1    = (const float*)d_in[8];
  const float* W2    = (const float*)d_in[9];
  const float* b2    = (const float*)d_in[10];
  const float* hidR  = (const float*)d_in[11];
  const float* hidI  = (const float*)d_in[12];
  const float* hidN  = (const float*)d_in[13];
  const float* inR   = (const float*)d_in[14];
  const float* inI   = (const float*)d_in[15];
  const float* inN   = (const float*)d_in[16];
  const float* o1W   = (const float*)d_in[17];
  const float* o1b   = (const float*)d_in[18];
  const float* o2W   = (const float*)d_in[19];
  const float* o2b   = (const float*)d_in[20];
  const float* o3W   = (const float*)d_in[21];
  const float* o3b   = (const float*)d_in[22];
  const float* aW    = (const float*)d_in[23];
  const float* ab    = (const float*)d_in[24];
  const float* gW    = (const float*)d_in[25];
  const float* gb    = (const float*)d_in[26];
  float* out = (float*)d_out;

  char* ws = (char*)d_ws;
  size_t off = 0;
  auto alloc = [&](size_t bytes) {
    char* p = ws + off;
    off = (off + bytes + 255) & ~(size_t)255;
    return p;
  };
  const size_t EN = (size_t)N_EDGES * N_NODES;
  __bf16* sendR_bf  = (__bf16*)alloc(EN * 2);
  __bf16* recvR_bf  = (__bf16*)alloc(EN * 2);
  __bf16* recvRT_bf = (__bf16*)alloc(EN * 2);
  __bf16* hidT_bf   = (__bf16*)alloc((size_t)HDIM * N_NODES * 2);
  __bf16* msg_pre   = (__bf16*)alloc((size_t)N_EDGES * HDIM2 * 2);
  __bf16* msgcT     = (__bf16*)alloc((size_t)HDIM * N_EDGES * 2);
  __bf16* maggbf    = (__bf16*)alloc((size_t)N_NODES * HDIM * 2);
  float*  agg_part  = (float*)alloc((size_t)KSPLIT * N_NODES * HDIM * 4);
  float*  hidden    = (float*)alloc((size_t)N_NODES * HDIM * 4);
  float*  prev_pred = (float*)alloc((size_t)N_NODES * 4);
  float*  scal      = (float*)alloc(256);
  __bf16* W1bf   = (__bf16*)alloc((size_t)HDIM * HDIM2 * 2);
  __bf16* W2bf   = (__bf16*)alloc((size_t)HDIM * HDIM * 2);
  __bf16* hidRbf = (__bf16*)alloc((size_t)HDIM * HDIM * 2);
  __bf16* hidIbf = (__bf16*)alloc((size_t)HDIM * HDIM * 2);
  __bf16* hidNbf = (__bf16*)alloc((size_t)HDIM * HDIM * 2);
  __bf16* o1Wbf  = (__bf16*)alloc((size_t)HDIM * HDIM * 2);
  __bf16* o2Wbf  = (__bf16*)alloc((size_t)HDIM * HDIM * 2);

  const int nEN = N_EDGES * N_NODES;
  const int nHH = HDIM * HDIM;
  k_cvt<<<(nEN + 255) / 256, 256, 0, stream>>>(sendR, sendR_bf, nEN);
  k_cvt<<<(nEN + 255) / 256, 256, 0, stream>>>(recvR, recvR_bf, nEN);
  k_transpose<<<(nEN + 255) / 256, 256, 0, stream>>>(recvR, recvRT_bf, N_EDGES, N_NODES);
  k_cvt<<<(HDIM * HDIM2 + 255) / 256, 256, 0, stream>>>(W1 + HDIM * HDIM2, W1bf, HDIM * HDIM2);
  k_cvt<<<(nHH + 255) / 256, 256, 0, stream>>>(W2 + nHH, W2bf, nHH);
  k_cvt<<<(nHH + 255) / 256, 256, 0, stream>>>(hidR, hidRbf, nHH);
  k_cvt<<<(nHH + 255) / 256, 256, 0, stream>>>(hidI, hidIbf, nHH);
  k_cvt<<<(nHH + 255) / 256, 256, 0, stream>>>(hidN, hidNbf, nHH);
  k_cvt<<<(nHH + 255) / 256, 256, 0, stream>>>(o1W, o1Wbf, nHH);
  k_cvt<<<(nHH + 255) / 256, 256, 0, stream>>>(o2W, o2Wbf, nHH);
  k_init<<<(N_NODES * HDIM + 255) / 256, 256, 0, stream>>>(
      hidden, hidT_bf, prev_pred, cont, aW, ab, gW, gb, scal);

  for (int t = 0; t < N_STEPS; ++t) {
    k_msgpre<<<dim3(4, N_EDGES / 16), 32, 0, stream>>>(sendR_bf, recvR_bf, hidT_bf, msg_pre);
    k_mlp<<<N_EDGES / 16, 256, 0, stream>>>(msg_pre, W1bf, W2bf, b1 + HDIM, b2 + HDIM,
                                            edges, msgcT);
    k_agg<<<dim3(HDIM / 32, N_NODES / 32, KSPLIT), 32, 0, stream>>>(recvRT_bf, msgcT, agg_part);
    k_agg_reduce<<<(N_NODES * HDIM + 255) / 256, 256, 0, stream>>>(agg_part, maggbf);
    k_gru_out<<<N_NODES / 16, 256, 0, stream>>>(
        maggbf, hidRbf, hidIbf, hidNbf, inR, inI, inN,
        o1Wbf, o1b, o2Wbf, o2b, o3W, o3b, data, tml,
        hidden, hidT_bf, prev_pred, scal, out, t);
  }
}
